// GroupQueryAttention_19756849562259
// MI455X (gfx1250) — matrix operand order repaired and checked by reading, not compiled, not run on hardware
//
#include <hip/hip_runtime.h>

// Problem constants (match reference)
#define BB 2
#define SS 2048
#define DD 1024
#define GG 8
#define HH 128            // DD / GG
#define BS (BB * SS)      // 4096 flattened rows
// scale = sqrt(D) = 32  ->  1/scale = 0.03125

typedef __attribute__((ext_vector_type(16))) __bf16 v16bf;
typedef __attribute__((ext_vector_type(8)))  float  v8f;

union FragBF { v16bf v; uint4 q[2]; };

__device__ __forceinline__ unsigned short cvt_bf16(float f) {
  unsigned u = __builtin_bit_cast(unsigned, f);
  u += 0x7FFFu + ((u >> 16) & 1u);          // round-to-nearest-even
  return (unsigned short)(u >> 16);
}
__device__ __forceinline__ unsigned pack2_bf16(float lo, float hi) {
  return (unsigned)cvt_bf16(lo) | ((unsigned)cvt_bf16(hi) << 16);
}
__device__ __forceinline__ v8f wmma_bf16(FragBF a, FragBF b, v8f c) {
  // D(16x16,f32) = A(16x32,bf16) * B(32x16,bf16) + C
  return __builtin_amdgcn_wmma_f32_16x16x32_bf16(false, a.v, false, b.v,
                                                 (short)0, c, false, false);
}
__device__ __forceinline__ v8f vzero8() {
  v8f z = {0.f, 0.f, 0.f, 0.f, 0.f, 0.f, 0.f, 0.f};
  return z;
}
// A-frag (16x32 bf16): lane reads 8 bf16 at K = 8*lh and 8 more at +16
__device__ __forceinline__ void ld_afrag(FragBF* f, const unsigned short* base) {
  f->q[0] = *(const uint4*)base;
  f->q[1] = *(const uint4*)(base + 16);
}
// B-frag (32x16 bf16): lane reads 8 bf16 at K = 8*lh and 8 more at +16
__device__ __forceinline__ void ld_bfrag(FragBF* f, const unsigned short* base) {
  f->q[0] = *(const uint4*)base;
  f->q[1] = *(const uint4*)(base + 16);
}

// ---------------------------------------------------------------------------
// K1a: X fp32 -> bf16 (row-major [BS][D]); 4 elements/thread
// ---------------------------------------------------------------------------
__global__ __launch_bounds__(256) void gqa_cvt_x(const float* __restrict__ x,
                                                 unsigned short* __restrict__ xb) {
  size_t i = ((size_t)blockIdx.x * blockDim.x + threadIdx.x) * 4;
  float4 f = *(const float4*)(x + i);
  unsigned long long p = (unsigned long long)pack2_bf16(f.x, f.y)
                       | ((unsigned long long)pack2_bf16(f.z, f.w) << 32);
  *(unsigned long long*)(xb + i) = p;
}

// ---------------------------------------------------------------------------
// K1b: W (fp32, [k][out]) -> Wt (bf16, [out][k])  (tiled LDS transpose)
// ---------------------------------------------------------------------------
__global__ __launch_bounds__(256) void gqa_cvt_wt(const float* __restrict__ w,
                                                  unsigned short* __restrict__ wt) {
  __shared__ float tile[32][33];
  const int tx = threadIdx.x & 31, ty = threadIdx.x >> 5;   // 32 x 8
  const int bx = blockIdx.x * 32, by = blockIdx.y * 32;
  for (int r = 0; r < 32; r += 8)
    tile[ty + r][tx] = w[(size_t)(by + ty + r) * DD + bx + tx];
  __syncthreads();
  for (int r = 0; r < 32; r += 8)
    wt[(size_t)(bx + ty + r) * DD + by + tx] = cvt_bf16(tile[tx][ty + r]);
}

// ---------------------------------------------------------------------------
// K2: QKV GEMM, one output matrix per blockIdx.z.
//   Per wave: 32 rows x 64 cols, K ping-pong double-buffered (step 64).
//   Q, K stored row-major bf16 [BS][D]; V stored transposed Vt[b][d][m].
// ---------------------------------------------------------------------------
__global__ __launch_bounds__(128) void gqa_qkv(
    const unsigned short* __restrict__ xb,
    const unsigned short* __restrict__ wtq,
    const unsigned short* __restrict__ wtk,
    const unsigned short* __restrict__ wtv,
    const float* __restrict__ bq, const float* __restrict__ bk,
    const float* __restrict__ bv,
    unsigned short* __restrict__ qout, unsigned short* __restrict__ kout,
    unsigned short* __restrict__ vtout)
{
  const int lane = threadIdx.x & 31, wave = threadIdx.x >> 5;
  const int ln = lane & 15, lh = lane >> 4;
  const int rt = blockIdx.x;                       // 32-row tile (0..127)
  const int col0 = (blockIdx.y * 4 + wave) * 64;   // 64-col chunk
  const int w = blockIdx.z;                        // 0:Q 1:K 2:V
  const unsigned short* wt = (w == 0) ? wtq : (w == 1) ? wtk : wtv;
  const float* bias = (w == 0) ? bq : (w == 1) ? bk : bv;

  v8f acc[2][4];
  for (int s = 0; s < 2; s++)
    for (int c = 0; c < 4; c++) acc[s][c] = vzero8();

  const unsigned short* ar0 = xb + (size_t)(rt * 32 + ln) * DD;
  const unsigned short* ar1 = ar0 + (size_t)16 * DD;
  const unsigned short* wrow[4];
  for (int c = 0; c < 4; c++)
    wrow[c] = wt + (size_t)(col0 + c * 16 + ln) * DD;

  FragBF a0[2], b0[4], a1[2], b1[4];
  auto loadAB = [&](FragBF a[2], FragBF b[4], int k0) {
    ld_afrag(&a[0], ar0 + k0 + 8 * lh);
    ld_afrag(&a[1], ar1 + k0 + 8 * lh);
    #pragma unroll
    for (int c = 0; c < 4; c++) ld_bfrag(&b[c], wrow[c] + k0 + 8 * lh);
  };
  auto doWmma = [&](FragBF a[2], FragBF b[4]) {
    #pragma unroll
    for (int s = 0; s < 2; s++)
      #pragma unroll
      for (int c = 0; c < 4; c++) acc[s][c] = wmma_bf16(a[s], b[c], acc[s][c]);
  };

  loadAB(a0, b0, 0);
  for (int k0 = 0; k0 < DD; k0 += 64) {
    if (k0 + 32 < DD) loadAB(a1, b1, k0 + 32);   // in flight over next 8 WMMAs
    doWmma(a0, b0);
    if (k0 + 64 < DD) loadAB(a0, b0, k0 + 64);
    doWmma(a1, b1);
  }

  // Epilogue: bias (fp32) + convert + store.  C layout: (M = r + 8*lh, N = ln)
  for (int c = 0; c < 4; c++) {
    const int col = col0 + c * 16 + ln;
    const float bs_ = bias[col];
    for (int s = 0; s < 2; s++) {
      for (int r = 0; r < 8; r++) {
        const int row = rt * 32 + s * 16 + r + 8 * lh;   // flattened b*S + s
        const unsigned short h = cvt_bf16(acc[s][c][r] + bs_);
        if (w == 0)      qout[(size_t)row * DD + col] = h;
        else if (w == 1) kout[(size_t)row * DD + col] = h;
        else {
          const int bidx = row >> 11, sq = row & (SS - 1);
          vtout[((size_t)bidx * DD + col) * SS + sq] = h;   // V transposed
        }
      }
    }
  }
}

// ---------------------------------------------------------------------------
// K3: Z[b,n,m] = sum_g exp(s_g[n,m] / 32), stored transposed Zt[b][m][n]
//   Wave = (b, n-tile16, m-chunk256).  Scores computed TRANSPOSED:
//   St[m,n] = sum_h K[m,h] * Q[n,h]   (A = K rows, B = Q^T)
//   All 24 frag loads of a (mtile,g) step batched before the 8 WMMAs.
// ---------------------------------------------------------------------------
__global__ __launch_bounds__(128) void gqa_zdenom(
    const unsigned short* __restrict__ qb, const unsigned short* __restrict__ kb,
    float* __restrict__ zt)
{
  const int lane = threadIdx.x & 31, wave = threadIdx.x >> 5;
  const int W = blockIdx.x * 4 + wave;             // 0..2047
  const int nt = W & 127, mc = (W >> 7) & 7, b = W >> 10;
  const int ln = lane & 15, lh = lane >> 4;
  const int n0 = nt * 16;

  const unsigned short* qrow = qb + (size_t)(b * SS + n0 + ln) * DD;  // B: N=n=ln
  const unsigned short* kbb  = kb + (size_t)b * SS * DD;
  float* ztb = zt + (size_t)b * SS * SS;

  for (int mt = 0; mt < 8; mt++) {
    const int m0 = mc * 256 + mt * 32;
    const unsigned short* krow0 = kbb + (size_t)(m0 + ln) * DD;
    const unsigned short* krow1 = krow0 + (size_t)16 * DD;
    v8f zsum0 = vzero8(), zsum1 = vzero8();
    for (int g = 0; g < GG; g++) {
      const int dg = g * HH;
      FragBF qf[4], kf0[4], kf1[4];
      #pragma unroll
      for (int kc = 0; kc < 4; kc++) {             // h = dg + kc*32 ..
        ld_bfrag(&qf[kc],  qrow  + dg + kc * 32 + 8 * lh);
        ld_afrag(&kf0[kc], krow0 + dg + kc * 32 + 8 * lh);
        ld_afrag(&kf1[kc], krow1 + dg + kc * 32 + 8 * lh);
      }
      v8f st0 = vzero8(), st1 = vzero8();
      #pragma unroll
      for (int kc = 0; kc < 4; kc++) {
        st0 = wmma_bf16(kf0[kc], qf[kc], st0);
        st1 = wmma_bf16(kf1[kc], qf[kc], st1);
      }
      #pragma unroll
      for (int r = 0; r < 8; r++) {
        zsum0[r] += __expf(st0[r] * 0.03125f);
        zsum1[r] += __expf(st1[r] * 0.03125f);
      }
    }
    // St C layout: (m = m0 + 16*sub + r + 8*lh, n = n0 + ln) -> lane-coalesced
    for (int r = 0; r < 8; r++) {
      ztb[(size_t)(m0 + r + 8 * lh) * SS + n0 + ln]      = zsum0[r];
      ztb[(size_t)(m0 + 16 + r + 8 * lh) * SS + n0 + ln] = zsum1[r];
    }
  }
}

// ---------------------------------------------------------------------------
// K4: ctx[b,n,g,:] = sum_m (exp(s_g/32)/Z) * V[m,g,:]
//   Wave = (b, g, n-tile16); loops all m.  St^T C-layout converts in-lane into
//   the bf16 A-fragment of P (lane = n, VGPR pairs = m) -- no LDS needed.
//   V-frag loads issued before the exp/normalize/pack block to hide latency.
// ---------------------------------------------------------------------------
__global__ __launch_bounds__(128) void gqa_ctx(
    const unsigned short* __restrict__ qb, const unsigned short* __restrict__ kb,
    const unsigned short* __restrict__ vtb, const float* __restrict__ zt,
    float* __restrict__ out)
{
  const int lane = threadIdx.x & 31, wave = threadIdx.x >> 5;
  const int W = blockIdx.x * 4 + wave;             // 0..2047
  const int nt = W & 127, g = (W >> 7) & 7, b = W >> 10;
  const int ln = lane & 15, lh = lane >> 4;
  const int n0 = nt * 16, dg = g * HH;

  // Hoist the 4 Q B-fragments for this (b, n-tile, g): 32 VGPRs
  const unsigned short* qrow = qb + (size_t)(b * SS + n0 + ln) * DD + dg;
  FragBF qf[4];
  #pragma unroll
  for (int kc = 0; kc < 4; kc++)
    ld_bfrag(&qf[kc], qrow + kc * 32 + 8 * lh);

  const unsigned short* kbb = kb  + (size_t)b * SS * DD;
  const unsigned short* vbb = vtb + (size_t)b * DD * SS;
  const float* ztb = zt + (size_t)b * SS * SS;

  v8f ctx[8];
  for (int c = 0; c < 8; c++) ctx[c] = vzero8();

  for (int m0 = 0; m0 < SS; m0 += 32) {
    if (m0 + 32 < SS) {  // prefetch next m-tile's K / V streams into cache
      __builtin_prefetch(kbb + (size_t)(m0 + 32 + ln) * DD + dg, 0, 1);
      __builtin_prefetch(vbb + (size_t)(dg + ln) * SS + m0 + 32, 0, 1);
    }
    // Scores (transposed), this group only: 16 batched loads then 8 WMMAs
    const unsigned short* krow0 = kbb + (size_t)(m0 + ln) * DD + dg;
    const unsigned short* krow1 = krow0 + (size_t)16 * DD;
    FragBF kf0[4], kf1[4];
    #pragma unroll
    for (int kc = 0; kc < 4; kc++) {
      ld_afrag(&kf0[kc], krow0 + kc * 32 + 8 * lh);
      ld_afrag(&kf1[kc], krow1 + kc * 32 + 8 * lh);
    }
    v8f st0 = vzero8(), st1 = vzero8();
    #pragma unroll
    for (int kc = 0; kc < 4; kc++) {
      st0 = wmma_bf16(kf0[kc], qf[kc], st0);
      st1 = wmma_bf16(kf1[kc], qf[kc], st1);
    }
    // Issue all V B-frag loads now; exp/normalize below hides their latency
    FragBF vf[8];
    #pragma unroll
    for (int c = 0; c < 8; c++)
      ld_bfrag(&vf[c], vbb + (size_t)(dg + c * 16 + ln) * SS + m0 + 8 * lh);
    // probs = exp(s/32) * rcp(Z)  (Z coalesced: lanes 0-15 consecutive n)
    float p0[8], p1[8];
    #pragma unroll
    for (int r = 0; r < 8; r++) {
      const float z0 = ztb[(size_t)(m0 + r + 8 * lh) * SS + n0 + ln];
      const float z1 = ztb[(size_t)(m0 + 16 + r + 8 * lh) * SS + n0 + ln];
      p0[r] = __expf(st0[r] * 0.03125f) * __builtin_amdgcn_rcpf(z0);
      p1[r] = __expf(st1[r] * 0.03125f) * __builtin_amdgcn_rcpf(z1);
    }
    // St^T C-layout -> P A-fragment: VGPR v = pack(p[2v], p[2v+1]); q[1] = sub1
    FragBF aP;
    aP.q[0] = make_uint4(pack2_bf16(p0[0], p0[1]), pack2_bf16(p0[2], p0[3]),
                         pack2_bf16(p0[4], p0[5]), pack2_bf16(p0[6], p0[7]));
    aP.q[1] = make_uint4(pack2_bf16(p1[0], p1[1]), pack2_bf16(p1[2], p1[3]),
                         pack2_bf16(p1[4], p1[5]), pack2_bf16(p1[6], p1[7]));
    // ctx += P(16x32) * V(32x128)
    #pragma unroll
    for (int c = 0; c < 8; c++)
      ctx[c] = wmma_bf16(aP, vf[c], ctx[c]);
  }
  // Store ctx fp32 (reference output dtype); lanes 0-15 contiguous in d
  for (int c = 0; c < 8; c++)
    for (int r = 0; r < 8; r++)
      out[(size_t)(b * SS + n0 + r + 8 * lh) * DD + dg + c * 16 + ln] = ctx[c][r];
}

// ---------------------------------------------------------------------------
extern "C" void kernel_launch(void* const* d_in, const int* in_sizes, int n_in,
                              void* d_out, int out_size, void* d_ws, size_t ws_size,
                              hipStream_t stream) {
  const float* x  = (const float*)d_in[0];
  const float* Wq = (const float*)d_in[1];
  const float* bq = (const float*)d_in[2];
  const float* Wk = (const float*)d_in[3];
  const float* bk = (const float*)d_in[4];
  const float* Wv = (const float*)d_in[5];
  const float* bv = (const float*)d_in[6];
  float* out = (float*)d_out;

  char* ws = (char*)d_ws;
  size_t off = 0;
  auto walloc = [&](size_t bytes) -> char* {
    char* p = ws + off;
    off += (bytes + 255) & ~(size_t)255;
    return p;
  };
  unsigned short* xb  = (unsigned short*)walloc((size_t)BS * DD * 2);  //  8 MB
  unsigned short* wtq = (unsigned short*)walloc((size_t)DD * DD * 2);  //  2 MB
  unsigned short* wtk = (unsigned short*)walloc((size_t)DD * DD * 2);  //  2 MB
  unsigned short* wtv = (unsigned short*)walloc((size_t)DD * DD * 2);  //  2 MB
  unsigned short* qbf = (unsigned short*)walloc((size_t)BS * DD * 2);  //  8 MB
  unsigned short* kbf = (unsigned short*)walloc((size_t)BS * DD * 2);  //  8 MB
  unsigned short* vtb = (unsigned short*)walloc((size_t)BS * DD * 2);  //  8 MB
  float*          ztd = (float*)walloc((size_t)BB * SS * SS * 4);      // 32 MB

  gqa_cvt_x<<<(BS * DD / 4) / 256, 256, 0, stream>>>(x, xb);
  dim3 tg(DD / 32, DD / 32);
  gqa_cvt_wt<<<tg, 256, 0, stream>>>(Wq, wtq);
  gqa_cvt_wt<<<tg, 256, 0, stream>>>(Wk, wtk);
  gqa_cvt_wt<<<tg, 256, 0, stream>>>(Wv, wtv);

  gqa_qkv<<<dim3(BS / 32, 4, 3), 128, 0, stream>>>(xb, wtq, wtk, wtv,
                                                   bq, bk, bv, qbf, kbf, vtb);
  gqa_zdenom<<<512, 128, 0, stream>>>(qbf, kbf, ztd);
  gqa_ctx<<<512, 128, 0, stream>>>(qbf, kbf, vtb, ztd, out);
}
